// DecoderAttention_62989990363726
// MI455X (gfx1250) — matrix operand order repaired and checked by reading, not compiled, not run on hardware
//
#include <hip/hip_runtime.h>

// Problem dims (fixed by the reference)
#define NB    256
#define NSRC  512
#define NPRED 128
#define ND    512

#define PT   32               // pred rows per workgroup
#define EC   64               // enc rows per streamed chunk
#define EFP  516              // f32 elems per enc row in LDS (pad: 516 % 64 == 4)
#define SP   528              // bf16 elems per score row in LDS (1056B stride, 16B aligned)

typedef __attribute__((ext_vector_type(16))) __bf16 v16bf;
typedef __attribute__((ext_vector_type(8)))  __bf16 v8bf;
typedef __attribute__((ext_vector_type(4)))  __bf16 v4bf;
typedef __attribute__((ext_vector_type(8)))  float  v8f;
typedef __attribute__((ext_vector_type(4)))  float  v4f;

static __device__ __forceinline__ __bf16 bfc(float x) { return (__bf16)x; }

// 16-byte global -> LDS copy. On gfx1250 use the async DMA path (no VGPR round
// trip, tracked by ASYNCcnt); elsewhere (incl. host pass) plain load+store.
static __device__ __forceinline__ void copy16_g2l(const float* g, float* l) {
#if defined(__gfx1250__)
  const unsigned int lds_off = (unsigned int)(unsigned long long)l; // addr[31:0]
  const unsigned long long gaddr = (unsigned long long)g;
  asm volatile("global_load_async_to_lds_b128 %0, %1, off"
               :
               : "v"(lds_off), "v"(gaddr)
               : "memory");
#else
  *(v4f*)l = *(const v4f*)g;
#endif
}

static __device__ __forceinline__ void async_wait_all() {
#if defined(__gfx1250__)
  asm volatile("s_wait_asynccnt 0" ::: "memory");
#endif
}

static __device__ __forceinline__ v8f wmma_bf16(v16bf a, v16bf b, v8f c) {
  // 8 args: (neg_a, A, neg_b, B, c_mod, C, reuse_a, reuse_b)
  return __builtin_amdgcn_wmma_f32_16x16x32_bf16(false, a, false, b, (short)0, c,
                                                 false, false);
}

// A fragment (16x32 bf16) from a row-major bf16 tile in LDS.
// lane(m = lane&15, half = lane>>4), element t -> k = half*16 + t
static __device__ __forceinline__ v16bf load_a_bf16(const __bf16* tile, int pitch,
                                                    int lane) {
  const int m = lane & 15, h = lane >> 4;
  const __bf16* r = tile + (size_t)m * pitch + h * 16;
  v8bf lo = *(const v8bf*)(r);       // k = h*16 + 0..7
  v8bf hi = *(const v8bf*)(r + 8);   // k = h*16 + 8..15
  v16bf a;
#pragma unroll
  for (int i = 0; i < 8; ++i) { a[i] = lo[i]; a[8 + i] = hi[i]; }
  return a;
}

// B fragment from 16 consecutive f32 (k = t + 16*half contiguous), convert to bf16.
static __device__ __forceinline__ v16bf load_b_f32_contig(const float* p) {
  v4f x0 = *(const v4f*)(p);
  v4f x1 = *(const v4f*)(p + 4);
  v4f x2 = *(const v4f*)(p + 8);
  v4f x3 = *(const v4f*)(p + 12);
  v16bf b;
#pragma unroll
  for (int i = 0; i < 4; ++i) {
    b[i]      = bfc(x0[i]);
    b[4 + i]  = bfc(x1[i]);
    b[8 + i]  = bfc(x2[i]);
    b[12 + i] = bfc(x3[i]);
  }
  return b;
}

__global__ __launch_bounds__(256, 1)
void decoder_attn(const float* __restrict__ enc,          // (B, SRC, D)
                  const float* __restrict__ dec,          // (B, PRED, D)
                  const unsigned char* __restrict__ mask, // (B, SRC) bool
                  const float* __restrict__ W,            // (D, D) torch (out,in)
                  float* __restrict__ out)                // (B, PRED, D)
{
  __shared__ __align__(16) __bf16 lds_s[PT * SP];      // scores/probs (bf16) 33.0KB
  __shared__ __align__(16) __bf16 lds_q[PT * ND];      // projected queries  32.0KB
  __shared__ __align__(16) float  lds_encf[EC * EFP];  // enc chunk (f32)   129.0KB
  __shared__ float lds_l[PT];                          // softmax row sums

  __bf16* lds_dec = lds_s;   // dec tile only lives before scores exist

  const int tid  = threadIdx.x;
  const int lane = tid & 31;
  const int wv   = tid >> 5;            // wave 0..7
  const int m16  = lane & 15;
  const int half = lane >> 4;

  const int b  = blockIdx.x;
  const int p0 = blockIdx.y * PT;

  const float* decB = dec + ((size_t)b * NPRED + p0) * ND;
  const float* encB = enc + (size_t)b * NSRC * ND;
  const unsigned char* maskB = mask + (size_t)b * NSRC;
  float* outB = out + ((size_t)b * NPRED + p0) * ND;

  // ---------- Phase 1: dec tile -> LDS (async f32 staging, then one cvt pass) ----
  float* stage = lds_encf;              // 64KB staging inside the enc buffer
  for (int i = tid; i < PT * ND / 4; i += 256)
    copy16_g2l(decB + i * 4, stage + i * 4);
  async_wait_all();
  __syncthreads();
  for (int i = tid; i < PT * ND / 4; i += 256) {
    v4f v = *(const v4f*)(stage + i * 4);
    *(v4bf*)(lds_dec + i * 4) = (v4bf){bfc(v[0]), bfc(v[1]), bfc(v[2]), bfc(v[3])};
  }
  __syncthreads();

  // ---------- Phase 2: Q = dec @ W^T -> lds_q (bf16) ----------
  // query[p,e] = sum_d dec[p,d] * W[e,d];  B[k=d][n=e] = W[e,d]
  for (int nn = 0; nn < 4; ++nn) {
    const int ni = wv * 4 + nn;                     // e-chunk 0..31
    v8f acc0 = {}; v8f acc1 = {};
    for (int kc = 0; kc < ND / 32; ++kc) {
      const int k0 = kc * 32;
      const float* wp = W + (size_t)(ni * 16 + m16) * ND + k0 + 16 * half;
      v16bf bb = load_b_f32_contig(wp);
      v16bf a0 = load_a_bf16(lds_dec + 0 * 16 * ND + k0, ND, lane);
      v16bf a1 = load_a_bf16(lds_dec + 1 * 16 * ND + k0, ND, lane);
      acc0 = wmma_bf16(a0, bb, acc0);
      acc1 = wmma_bf16(a1, bb, acc1);
    }
#pragma unroll
    for (int j = 0; j < 8; ++j) {   // C layout: m = j + 8*half, n = m16
      lds_q[(size_t)(0 * 16 + j + 8 * half) * ND + ni * 16 + m16] = bfc(acc0[j]);
      lds_q[(size_t)(1 * 16 + j + 8 * half) * ND + ni * 16 + m16] = bfc(acc1[j]);
    }
  }
  __syncthreads();

  // ---------- Phase 3: S = Q @ enc^T (async-streamed enc chunks) ----------
  for (int c = 0; c < NSRC / EC; ++c) {
    const float* src = encB + (size_t)c * EC * ND;
    for (int i = tid; i < EC * (ND / 4); i += 256) {
      const int r = i >> 7, cc = (i & 127) * 4;      // ND/4 == 128
      copy16_g2l(src + (size_t)r * ND + cc, lds_encf + (size_t)r * EFP + cc);
    }
    if (c + 1 < NSRC / EC) {                          // warm L2 for next chunk
      const float* nxt = encB + (size_t)(c + 1) * EC * ND;
      for (int i = tid * 32; i < EC * ND; i += 256 * 32)
        __builtin_prefetch(nxt + i, 0, 0);
    }
    async_wait_all();
    __syncthreads();

    // one 16x16 S-tile per wave: rows mi*16.., cols c*64 + ni*16..
    {
      const int mi = wv & 1, ni = wv >> 1;
      v8f acc = {};
      for (int kc = 0; kc < ND / 32; ++kc) {
        v16bf a = load_a_bf16(lds_q + (size_t)mi * 16 * ND + kc * 32, ND, lane);
        // B[k=d][n=s] = enc[s,d]: contiguous along d in f32 LDS row s
        const float* ep = lds_encf + (size_t)(ni * 16 + m16) * EFP + kc * 32 + 16 * half;
        acc = wmma_bf16(a, load_b_f32_contig(ep), acc);
      }
#pragma unroll
      for (int j = 0; j < 8; ++j)
        lds_s[(size_t)(mi * 16 + j + 8 * half) * SP + c * EC + ni * 16 + m16] =
            bfc(acc[j]);
    }
    __syncthreads();   // protect lds_encf reuse next iteration
  }

  // ---------- Phase 4: mask + row softmax (probs in bf16, sums in lds_l) -------
  for (int r = wv; r < PT; r += 8) {
    float vals[NSRC / 32];
    float mx = -__builtin_inff();
#pragma unroll
    for (int i = 0; i < NSRC / 32; ++i) {
      const int s = i * 32 + lane;
      const float sv =
          maskB[s] ? -__builtin_inff() : (float)lds_s[(size_t)r * SP + s];
      vals[i] = sv;
      mx = fmaxf(mx, sv);
    }
#pragma unroll
    for (int off = 16; off > 0; off >>= 1)
      mx = fmaxf(mx, __shfl_xor(mx, off, 32));
    float sum = 0.f;
#pragma unroll
    for (int i = 0; i < NSRC / 32; ++i) {
      const float p = (vals[i] == -__builtin_inff()) ? 0.f : __expf(vals[i] - mx);
      sum += p;
      lds_s[(size_t)r * SP + i * 32 + lane] = bfc(p);
    }
#pragma unroll
    for (int off = 16; off > 0; off >>= 1)
      sum += __shfl_xor(sum, off, 32);
    if (lane == 0) lds_l[r] = sum;
  }
  __syncthreads();

  // ---------- Phase 5: O = P @ enc (re-stream enc chunks, accumulate) ----------
  v8f o[4][2];
#pragma unroll
  for (int nn = 0; nn < 4; ++nn) { o[nn][0] = (v8f){}; o[nn][1] = (v8f){}; }

  for (int c = 0; c < NSRC / EC; ++c) {
    const float* src = encB + (size_t)c * EC * ND;
    for (int i = tid; i < EC * (ND / 4); i += 256) {
      const int r = i >> 7, cc = (i & 127) * 4;
      copy16_g2l(src + (size_t)r * ND + cc, lds_encf + (size_t)r * EFP + cc);
    }
    if (c + 1 < NSRC / EC) {
      const float* nxt = encB + (size_t)(c + 1) * EC * ND;
      for (int i = tid * 32; i < EC * ND; i += 256 * 32)
        __builtin_prefetch(nxt + i, 0, 0);
    }
    async_wait_all();
    __syncthreads();

    for (int kc2 = 0; kc2 < 2; ++kc2) {
      const int kbase = c * EC + kc2 * 32;   // K (=s) base of this step
      v16bf a0 = load_a_bf16(lds_s + (size_t)0 * 16 * SP + kbase, SP, lane);
      v16bf a1 = load_a_bf16(lds_s + (size_t)1 * 16 * SP + kbase, SP, lane);
#pragma unroll
      for (int nn = 0; nn < 4; ++nn) {
        const int ni = wv * 4 + nn;          // d-chunk 0..31
        // B[k=s][n=d] = enc[s,d]: column walk (EFP % 64 == 4 -> conflict-free)
        const float* eb =
            lds_encf + (size_t)(kc2 * 32 + 16 * half) * EFP + ni * 16 + m16;
        v16bf bb;
#pragma unroll
        for (int t = 0; t < 16; ++t) bb[t] = bfc(eb[(size_t)t * EFP]);
        o[nn][0] = wmma_bf16(a0, bb, o[nn][0]);
        o[nn][1] = wmma_bf16(a1, bb, o[nn][1]);
      }
    }
    __syncthreads();
  }

  // ---------- normalize by softmax row sums and store ----------
#pragma unroll
  for (int nn = 0; nn < 4; ++nn) {
    const int ni = wv * 4 + nn;
#pragma unroll
    for (int j = 0; j < 8; ++j) {
      const int r0 = 0 * 16 + j + 8 * half;
      const int r1 = 1 * 16 + j + 8 * half;
      outB[(size_t)r0 * ND + ni * 16 + m16] = o[nn][0][j] / lds_l[r0];
      outB[(size_t)r1 * ND + ni * 16 + m16] = o[nn][1][j] / lds_l[r1];
    }
  }
}

extern "C" void kernel_launch(void* const* d_in, const int* in_sizes, int n_in,
                              void* d_out, int out_size, void* d_ws, size_t ws_size,
                              hipStream_t stream) {
  const float*         enc  = (const float*)d_in[0];          // (B,SRC,D) f32
  const float*         dec  = (const float*)d_in[1];          // (B,PRED,D) f32
  const unsigned char* mask = (const unsigned char*)d_in[2];  // (B,SRC) bool
  const float*         W    = (const float*)d_in[3];          // (D,D) f32
  float* out = (float*)d_out;                                 // (B,PRED,D) f32

  dim3 grid(NB, NPRED / PT);   // 256 x 4 workgroups
  decoder_attn<<<grid, 256, 0, stream>>>(enc, dec, mask, W, out);

  (void)in_sizes; (void)n_in; (void)out_size; (void)d_ws; (void)ws_size;
}
